// TransformerDecoder_31138512896463
// MI455X (gfx1250) — matrix operand order repaired and checked by reading, not compiled, not run on hardware
//
#include <hip/hip_runtime.h>
#include <hip/hip_bf16.h>
#include <math.h>
#include <stdint.h>

typedef _Float16 v16h __attribute__((ext_vector_type(16)));
typedef _Float16 v8h  __attribute__((ext_vector_type(8)));
typedef float    v8f  __attribute__((ext_vector_type(8)));

#define VOC   100277
#define EMB   768
#define NH    12
#define HD    64
#define NL    6
#define SEQ   256
#define NB    4
#define MROWS 1024     // NB*SEQ
#define FFD   3072

// ---------------------------------------------------------------------------
// Tiled WMMA GEMM: C[M,N] = A[M,K](f16) * B[K,N] (+bias)(+res)(relu?)
// Block tile 128x128, K-step 32, 8 waves (2x4), wave tile 64x32 (4x2 frags).
// Double-buffered LDS + register-staged b128 copies (software pipelined).
// BT:   B accessed transposed (B[k][n] = Bmem[n*ldb + k])  -- for Q*K^T
// B32:  B is fp32 in memory, converted to f16 into LDS      -- for lm_w
// Batched via blockIdx.z with outer/inner strides (z = zo*ziCnt + zi).
// ---------------------------------------------------------------------------
template<bool BT, bool B32, bool BIAS, bool RELU, bool RES, bool ST32, bool ST16>
__global__ __launch_bounds__(256) void gemm128(
    const _Float16* __restrict__ A, const _Float16* __restrict__ Bh,
    const float* __restrict__ Bf, const float* __restrict__ bias,
    const float* __restrict__ res, float* __restrict__ out32,
    _Float16* __restrict__ out16,
    int M, int N, int K, int lda, int ldb, int ldc,
    long sAo, long sAi, long sBo, long sBi, long sCo, long sCi, int ziCnt)
{
  __shared__ alignas(16) _Float16 As[2][128][32];
  __shared__ alignas(16) _Float16 BsT[2][128][32];   // [n][k]

  const int tid  = threadIdx.x;
  const int lane = tid & 31;
  const int wave = tid >> 5;
  const int wm   = (wave >> 2) * 64;   // wave M origin in block tile
  const int wn   = (wave & 3) * 32;    // wave N origin
  const int l15  = lane & 15;
  const int lhi  = lane >> 4;          // 0 or 1

  const int zo = blockIdx.z / ziCnt;
  const int zi = blockIdx.z % ziCnt;
  const long offA = (long)zo * sAo + (long)zi * sAi;
  const long offB = (long)zo * sBo + (long)zi * sBi;
  const long offC = (long)zo * sCo + (long)zi * sCi;

  const int bm = blockIdx.y * 128;
  const int bn = blockIdx.x * 128;

  const _Float16* Abase = A + offA + (long)bm * lda;

  // per-thread copy coordinates (compile-time shapes, hoisted out of the loop)
  // A: 512 chunks of 8 halfs; chunk ch -> row ch>>2, col ((ch&3)<<3)
  const int aR0 = tid >> 2,           aC0 = (tid & 3) << 3;
  const int aR1 = (tid + 256) >> 2,   aC1 = ((tid + 256) & 3) << 3;
  // B (normal/B32): chunk ch -> k-row ch>>4, n0 ((ch&15)<<3)
  const int bK0 = tid >> 4,           bN0 = (tid & 15) << 3;
  const int bK1 = (tid + 256) >> 4,   bN1 = ((tid + 256) & 15) << 3;
  // B (BT): chunk ch -> n ch>>2, k0 ((ch&3)<<3)
  const int tN0 = tid >> 2,           tC0 = (tid & 3) << 3;
  const int tN1 = (tid + 256) >> 2,   tC1 = ((tid + 256) & 3) << 3;

  v8h   aReg[2];
  v8h   bReg[2];
  float bRegF[16];

  auto loadTiles = [&](int k0) {
    // ---- A tile (dims always multiples of 128/32 -> no guards)
    const _Float16* Ag = Abase + k0;
    aReg[0] = *(const v8h*)&Ag[(long)aR0 * lda + aC0];
    aReg[1] = *(const v8h*)&Ag[(long)aR1 * lda + aC1];
    // ---- B tile
    if (B32) {
      const float* Bg = Bf + offB + (long)k0 * ldb + bn;
      #pragma unroll
      for (int j = 0; j < 8; ++j) {
        bRegF[j]     = (bn + bN0 + j < N) ? Bg[(long)bK0 * ldb + bN0 + j] : 0.0f;
        bRegF[8 + j] = (bn + bN1 + j < N) ? Bg[(long)bK1 * ldb + bN1 + j] : 0.0f;
      }
    } else if (BT) {
      const _Float16* Bg = Bh + offB + (long)bn * ldb + k0;
      if (bn + tN0 < N) {
        bReg[0] = *(const v8h*)&Bg[(long)tN0 * ldb + tC0];
      } else {
        #pragma unroll
        for (int j = 0; j < 8; ++j) bReg[0][j] = (_Float16)0.0f;
      }
      if (bn + tN1 < N) {
        bReg[1] = *(const v8h*)&Bg[(long)tN1 * ldb + tC1];
      } else {
        #pragma unroll
        for (int j = 0; j < 8; ++j) bReg[1][j] = (_Float16)0.0f;
      }
    } else {
      const _Float16* Bg = Bh + offB + (long)k0 * ldb + bn;
      if (bn + bN0 + 8 <= N) {
        bReg[0] = *(const v8h*)&Bg[(long)bK0 * ldb + bN0];
      } else {
        #pragma unroll
        for (int j = 0; j < 8; ++j)
          bReg[0][j] = (bn + bN0 + j < N) ? Bg[(long)bK0 * ldb + bN0 + j]
                                          : (_Float16)0.0f;
      }
      if (bn + bN1 + 8 <= N) {
        bReg[1] = *(const v8h*)&Bg[(long)bK1 * ldb + bN1];
      } else {
        #pragma unroll
        for (int j = 0; j < 8; ++j)
          bReg[1][j] = (bn + bN1 + j < N) ? Bg[(long)bK1 * ldb + bN1 + j]
                                          : (_Float16)0.0f;
      }
    }
  };

  auto storeTiles = [&](int buf) {
    *(v8h*)&As[buf][aR0][aC0] = aReg[0];
    *(v8h*)&As[buf][aR1][aC1] = aReg[1];
    if (B32) {
      #pragma unroll
      for (int j = 0; j < 8; ++j) {
        BsT[buf][bN0 + j][bK0] = (_Float16)bRegF[j];
        BsT[buf][bN1 + j][bK1] = (_Float16)bRegF[8 + j];
      }
    } else if (BT) {
      *(v8h*)&BsT[buf][tN0][tC0] = bReg[0];
      *(v8h*)&BsT[buf][tN1][tC1] = bReg[1];
    } else {
      #pragma unroll
      for (int j = 0; j < 8; ++j) {
        BsT[buf][bN0 + j][bK0] = bReg[0][j];
        BsT[buf][bN1 + j][bK1] = bReg[1][j];
      }
    }
  };

  v8f acc[4][2];
  #pragma unroll
  for (int mi = 0; mi < 4; ++mi)
    #pragma unroll
    for (int ni = 0; ni < 2; ++ni)
      #pragma unroll
      for (int j = 0; j < 8; ++j) acc[mi][ni][j] = 0.0f;

  const int nk = K >> 5;             // K / 32 (always exact)
  loadTiles(0);
  storeTiles(0);
  __syncthreads();

  for (int ki = 0; ki < nk; ++ki) {
    const int cur = ki & 1;
    if (ki + 1 < nk) loadTiles((ki + 1) << 5);   // global loads in flight

    // ---- fragments (layouts per CDNA5 ISA 7.12.2, wave32)
    v16h af[4], bf2[2];
    #pragma unroll
    for (int mi = 0; mi < 4; ++mi) {
      int row = wm + mi * 16 + l15;
      v8h lo = *(const v8h*)&As[cur][row][lhi * 8];       // K = kb..kb+7
      v8h hi = *(const v8h*)&As[cur][row][16 + lhi * 8];  // K = 16+kb..
      #pragma unroll
      for (int j = 0; j < 8; ++j) { af[mi][j] = lo[j]; af[mi][8 + j] = hi[j]; }
    }
    #pragma unroll
    for (int ni = 0; ni < 2; ++ni) {
      int col = wn + ni * 16 + l15;
      v8h lo = *(const v8h*)&BsT[cur][col][lhi * 8];
      v8h hi = *(const v8h*)&BsT[cur][col][16 + lhi * 8];
      #pragma unroll
      for (int j = 0; j < 8; ++j) { bf2[ni][j] = lo[j]; bf2[ni][8 + j] = hi[j]; }
    }
    #pragma unroll
    for (int mi = 0; mi < 4; ++mi)
      #pragma unroll
      for (int ni = 0; ni < 2; ++ni)
        acc[mi][ni] = __builtin_amdgcn_wmma_f32_16x16x32_f16(
            false, af[mi], false, bf2[ni], (short)0, acc[mi][ni], false, false);

    if (ki + 1 < nk) storeTiles(1 - cur);        // fill the other buffer
    __syncthreads();
  }

  // ---- epilogue: C/D layout row = r + 8*(lane>=16), col = lane&15
  #pragma unroll
  for (int mi = 0; mi < 4; ++mi) {
    #pragma unroll
    for (int ni = 0; ni < 2; ++ni) {
      int gcol = bn + wn + ni * 16 + l15;
      if (gcol < N) {
        #pragma unroll
        for (int r = 0; r < 8; ++r) {
          int grow = bm + wm + mi * 16 + r + (lhi << 3);
          long oidx = offC + (long)grow * ldc + gcol;
          float v = acc[mi][ni][r];
          if (BIAS) v += bias[gcol];
          if (RES)  v += res[oidx];
          if (RELU) v = fmaxf(v, 0.0f);
          if (ST32) out32[oidx] = v;
          if (ST16) out16[oidx] = (_Float16)v;
        }
      }
    }
  }
}

// ---------------------------------------------------------------------------
__global__ __launch_bounds__(256) void embed_kernel(
    const int* __restrict__ idx, const float* __restrict__ tok,
    const float* __restrict__ pos, float* __restrict__ X)
{
  int row = blockIdx.x;
  int t = row & (SEQ - 1);
  int token = idx[row];
  for (int e = threadIdx.x; e < EMB; e += 256)
    X[(long)row * EMB + e] = tok[(long)token * EMB + e] + pos[(long)t * EMB + e];
}

__global__ __launch_bounds__(256) void ln_kernel(
    const float* __restrict__ x, const float* __restrict__ g,
    const float* __restrict__ b, _Float16* __restrict__ out)
{
  __shared__ float red[256];
  const int row = blockIdx.x, tid = threadIdx.x;
  const float* xr = x + (long)row * EMB;
  float s = 0.0f;
  for (int e = tid; e < EMB; e += 256) s += xr[e];
  red[tid] = s; __syncthreads();
  for (int st = 128; st > 0; st >>= 1) { if (tid < st) red[tid] += red[tid + st]; __syncthreads(); }
  const float mu = red[0] / (float)EMB; __syncthreads();
  float s2 = 0.0f;
  for (int e = tid; e < EMB; e += 256) { float d = xr[e] - mu; s2 += d * d; }
  red[tid] = s2; __syncthreads();
  for (int st = 128; st > 0; st >>= 1) { if (tid < st) red[tid] += red[tid + st]; __syncthreads(); }
  const float rstd = rsqrtf(red[0] / (float)EMB + 1e-5f);
  for (int e = tid; e < EMB; e += 256)
    out[(long)row * EMB + e] = (_Float16)((xr[e] - mu) * rstd * g[e] + b[e]);
}

__global__ __launch_bounds__(256) void softmax_kernel(
    const float* __restrict__ S, _Float16* __restrict__ P)
{
  __shared__ float red[256];
  const int row = blockIdx.x;               // over NB*NH*SEQ
  const int t = row & (SEQ - 1);
  const int tid = threadIdx.x;              // s index
  const float scale = 0.03608439182435161f; // E^-0.5 (reference scales by n_embed)
  const float* sr = S + (long)row * SEQ;
  float val = (tid <= t) ? sr[tid] * scale : -3.4e38f;
  red[tid] = val; __syncthreads();
  for (int st = 128; st > 0; st >>= 1) { if (tid < st) red[tid] = fmaxf(red[tid], red[tid + st]); __syncthreads(); }
  const float m = red[0]; __syncthreads();
  float e = (tid <= t) ? __expf(val - m) : 0.0f;
  red[tid] = e; __syncthreads();
  for (int st = 128; st > 0; st >>= 1) { if (tid < st) red[tid] += red[tid + st]; __syncthreads(); }
  P[(long)row * SEQ + tid] = (_Float16)(e / red[0]);
}

__global__ __launch_bounds__(256) void cvt_f32_f16_kernel(
    const float* __restrict__ in, _Float16* __restrict__ out, int n)
{
  int i = blockIdx.x * 256 + threadIdx.x;
  if (i < n) out[i] = (_Float16)in[i];
}

// (H,E,D) fp32 -> (E, H*D) f16  (so QKV is one plain row-major GEMM weight)
__global__ __launch_bounds__(256) void cvt_head_kernel(
    const float* __restrict__ w, _Float16* __restrict__ out)
{
  int i = blockIdx.x * 256 + threadIdx.x;
  if (i < EMB * EMB) {
    int e = i / EMB, c = i % EMB;
    int h = c >> 6, d = c & 63;
    out[i] = (_Float16)w[(long)h * EMB * HD + (long)e * HD + d];
  }
}

__global__ __launch_bounds__(256) void loss_rows_kernel(
    const float* __restrict__ logits, const int* __restrict__ tgt,
    float* __restrict__ rl)
{
  __shared__ float red[256];
  const int row = blockIdx.x, tid = threadIdx.x;
  const float* lr = logits + (long)row * VOC;
  float m = -3.4e38f;
  for (int v = tid; v < VOC; v += 256) m = fmaxf(m, lr[v]);
  red[tid] = m; __syncthreads();
  for (int st = 128; st > 0; st >>= 1) { if (tid < st) red[tid] = fmaxf(red[tid], red[tid + st]); __syncthreads(); }
  m = red[0]; __syncthreads();
  float s = 0.0f;
  for (int v = tid; v < VOC; v += 256) s += __expf(lr[v] - m);
  red[tid] = s; __syncthreads();
  for (int st = 128; st > 0; st >>= 1) { if (tid < st) red[tid] += red[tid + st]; __syncthreads(); }
  if (tid == 0) {
    float lse = m + logf(red[0]);
    rl[row] = lse - lr[tgt[row]];
  }
}

__global__ __launch_bounds__(256) void loss_reduce_kernel(
    const float* __restrict__ rl, float* __restrict__ out)
{
  __shared__ float red[256];
  const int tid = threadIdx.x;
  float s = 0.0f;
  for (int i = tid; i < MROWS; i += 256) s += rl[i];
  red[tid] = s; __syncthreads();
  for (int st = 128; st > 0; st >>= 1) { if (tid < st) red[tid] += red[tid + st]; __syncthreads(); }
  if (tid == 0) out[0] = red[0] / (float)MROWS;
}

// ---------------------------------------------------------------------------
extern "C" void kernel_launch(void* const* d_in, const int* in_sizes, int n_in,
                              void* d_out, int out_size, void* d_ws, size_t ws_size,
                              hipStream_t stream)
{
  (void)in_sizes; (void)n_in; (void)out_size; (void)ws_size;
  const int*   idx  = (const int*)d_in[0];
  const int*   tgt  = (const int*)d_in[1];
  const float* tok  = (const float*)d_in[2];
  const float* pos  = (const float*)d_in[3];
  const float* ln1g = (const float*)d_in[4];
  const float* ln1b = (const float*)d_in[5];
  const float* wq   = (const float*)d_in[6];
  const float* wk   = (const float*)d_in[7];
  const float* wv   = (const float*)d_in[8];
  const float* wo   = (const float*)d_in[9];
  const float* bo   = (const float*)d_in[10];
  const float* ln2g = (const float*)d_in[11];
  const float* ln2b = (const float*)d_in[12];
  const float* w1   = (const float*)d_in[13];
  const float* b1   = (const float*)d_in[14];
  const float* w2   = (const float*)d_in[15];
  const float* b2   = (const float*)d_in[16];
  const float* lnfg = (const float*)d_in[17];
  const float* lnfb = (const float*)d_in[18];
  const float* lmw  = (const float*)d_in[19];
  const float* lmb  = (const float*)d_in[20];

  float* logits  = (float*)d_out;
  float* lossOut = logits + (size_t)MROWS * VOC;

  char* wp = (char*)d_ws;
  auto carve = [&](size_t bytes) -> void* {
    void* p = (void*)wp; wp += (bytes + 255) & ~(size_t)255; return p;
  };
  float*    X   = (float*)   carve((size_t)MROWS * EMB * 4);
  _Float16* HLN = (_Float16*)carve((size_t)MROWS * EMB * 2);
  _Float16* Qb  = (_Float16*)carve((size_t)MROWS * EMB * 2);
  _Float16* Kb  = (_Float16*)carve((size_t)MROWS * EMB * 2);
  _Float16* Vb  = (_Float16*)carve((size_t)MROWS * EMB * 2);
  float*    Sb  = (float*)   carve((size_t)NB * NH * SEQ * SEQ * 4);
  _Float16* Pb  = (_Float16*)carve((size_t)NB * NH * SEQ * SEQ * 2);
  _Float16* Ob  = (_Float16*)carve((size_t)MROWS * EMB * 2);
  _Float16* Fb  = (_Float16*)carve((size_t)MROWS * FFD * 2);
  _Float16* WQc = (_Float16*)carve((size_t)EMB * EMB * 2);
  _Float16* WKc = (_Float16*)carve((size_t)EMB * EMB * 2);
  _Float16* WVc = (_Float16*)carve((size_t)EMB * EMB * 2);
  _Float16* WOc = (_Float16*)carve((size_t)EMB * EMB * 2);
  _Float16* W1c = (_Float16*)carve((size_t)EMB * FFD * 2);
  _Float16* W2c = (_Float16*)carve((size_t)EMB * FFD * 2);
  float*    RL  = (float*)   carve((size_t)MROWS * 4);

  const dim3 blk(256);
  const long sQo = (long)SEQ * EMB;           // per-batch stride in q/k/v/o (f16)
  const long sSo = (long)NH * SEQ * SEQ;      // per-batch stride in S/P

  embed_kernel<<<MROWS, blk, 0, stream>>>(idx, tok, pos, X);

  for (int l = 0; l < NL; ++l) {
    ln_kernel<<<MROWS, blk, 0, stream>>>(X, ln1g + l * EMB, ln1b + l * EMB, HLN);

    const int nEE = EMB * EMB, gEE = (nEE + 255) / 256;
    cvt_head_kernel<<<gEE, blk, 0, stream>>>(wq + (size_t)l * NH * EMB * HD, WQc);
    cvt_head_kernel<<<gEE, blk, 0, stream>>>(wk + (size_t)l * NH * EMB * HD, WKc);
    cvt_head_kernel<<<gEE, blk, 0, stream>>>(wv + (size_t)l * NH * EMB * HD, WVc);

    // Q/K/V = HLN @ W  (M=1024,N=768,K=768) -> f16 [b][t][h][d]
    gemm128<false,false,false,false,false,false,true><<<dim3(6,8,1), blk, 0, stream>>>(
        HLN, WQc, nullptr, nullptr, nullptr, nullptr, Qb,
        MROWS, EMB, EMB, EMB, EMB, EMB, 0,0,0,0,0,0, 1);
    gemm128<false,false,false,false,false,false,true><<<dim3(6,8,1), blk, 0, stream>>>(
        HLN, WKc, nullptr, nullptr, nullptr, nullptr, Kb,
        MROWS, EMB, EMB, EMB, EMB, EMB, 0,0,0,0,0,0, 1);
    gemm128<false,false,false,false,false,false,true><<<dim3(6,8,1), blk, 0, stream>>>(
        HLN, WVc, nullptr, nullptr, nullptr, nullptr, Vb,
        MROWS, EMB, EMB, EMB, EMB, EMB, 0,0,0,0,0,0, 1);

    // S[b,h,t,s] = q @ k^T   (batched over 48 (b,h); B transposed)
    gemm128<true,false,false,false,false,true,false><<<dim3(2,2,NB*NH), blk, 0, stream>>>(
        Qb, Kb, nullptr, nullptr, nullptr, Sb, nullptr,
        SEQ, SEQ, HD, EMB, EMB, SEQ,
        sQo, (long)HD, sQo, (long)HD, sSo, (long)SEQ * SEQ, NH);

    softmax_kernel<<<NB * NH * SEQ, blk, 0, stream>>>(Sb, Pb);

    // O[b,t,h*64+d] = P @ V
    gemm128<false,false,false,false,false,false,true><<<dim3(1,2,NB*NH), blk, 0, stream>>>(
        Pb, Vb, nullptr, nullptr, nullptr, nullptr, Ob,
        SEQ, HD, SEQ, SEQ, EMB, EMB,
        sSo, (long)SEQ * SEQ, sQo, (long)HD, sQo, (long)HD, NH);

    // x = x + O @ wo + bo
    cvt_f32_f16_kernel<<<gEE, blk, 0, stream>>>(wo + (size_t)l * EMB * EMB, WOc, nEE);
    gemm128<false,false,true,false,true,true,false><<<dim3(6,8,1), blk, 0, stream>>>(
        Ob, WOc, nullptr, bo + l * EMB, X, X, nullptr,
        MROWS, EMB, EMB, EMB, EMB, EMB, 0,0,0,0,0,0, 1);

    // FFN
    ln_kernel<<<MROWS, blk, 0, stream>>>(X, ln2g + l * EMB, ln2b + l * EMB, HLN);
    const int nEF = EMB * FFD, gEF = (nEF + 255) / 256;
    cvt_f32_f16_kernel<<<gEF, blk, 0, stream>>>(w1 + (size_t)l * EMB * FFD, W1c, nEF);
    gemm128<false,false,true,true,false,false,true><<<dim3(24,8,1), blk, 0, stream>>>(
        HLN, W1c, nullptr, b1 + l * FFD, nullptr, nullptr, Fb,
        MROWS, FFD, EMB, EMB, FFD, FFD, 0,0,0,0,0,0, 1);
    cvt_f32_f16_kernel<<<gEF, blk, 0, stream>>>(w2 + (size_t)l * FFD * EMB, W2c, nEF);
    gemm128<false,false,true,false,true,true,false><<<dim3(6,8,1), blk, 0, stream>>>(
        Fb, W2c, nullptr, b2 + l * EMB, X, X, nullptr,
        MROWS, EMB, FFD, FFD, EMB, EMB, 0,0,0,0,0,0, 1);
  }

  // final LN + LM head (B fp32 converted on the fly) + loss
  ln_kernel<<<MROWS, blk, 0, stream>>>(X, lnfg, lnfb, HLN);
  gemm128<false,true,true,false,false,true,false><<<dim3((VOC+127)/128, 8, 1), blk, 0, stream>>>(
      HLN, nullptr, lmw, lmb, nullptr, logits, nullptr,
      MROWS, VOC, EMB, EMB, VOC, VOC, 0,0,0,0,0,0, 1);

  loss_rows_kernel<<<MROWS, blk, 0, stream>>>(logits, tgt, RL);
  loss_reduce_kernel<<<1, blk, 0, stream>>>(RL, lossOut);
}
